// S_Att_41583873360517
// MI455X (gfx1250) — matrix operand order repaired and checked by reading, not compiled, not run on hardware
//
#include <hip/hip_runtime.h>
#include <hip/hip_bf16.h>
#include <math.h>

// Problem constants (match reference)
#define L_SEQ 4096
#define CDIM  1024
#define HEADS 16
#define DHEAD 64
#define HD    1024          // HEADS*DHEAD
#define LN_EPS 1e-5f

typedef __bf16 bf16;
typedef __attribute__((ext_vector_type(16))) __bf16 v16bf;
typedef __attribute__((ext_vector_type(8)))  float  v8f;
typedef __attribute__((ext_vector_type(4)))  unsigned int u32x4;
typedef __attribute__((address_space(3))) bf16 lds_bf16_t;

// ---------------------------------------------------------------------------
// WMMA helpers (CDNA5 wave32, D = A[16x32] * B[32x16] + C, fp32 accum)
// ---------------------------------------------------------------------------
__device__ __forceinline__ v8f wmma_bf16(v16bf a, v16bf b, v8f c) {
  // (neg_a, A, neg_b, B, c_mod, C, reuse_a, reuse_b)
  return __builtin_amdgcn_wmma_f32_16x16x32_bf16(false, a, false, b,
                                                 (short)0, c, false, false);
}

// A-matrix 16x32 bf16 fragment: lane holds row m=(lane&15); K chunks
// [kb, kb+8) and [kb+16, kb+24) where kb = 8*(lane>>4).  rowp points at K=0.
__device__ __forceinline__ v16bf load_a_frag(const bf16* rowp, int kb) {
  v16bf f;
  ((u32x4*)&f)[0] = *(const u32x4*)(rowp + kb);
  ((u32x4*)&f)[1] = *(const u32x4*)(rowp + kb + 16);
  return f;
}

// B-matrix 32x16 bf16 fragment: lane holds col n=(lane&15); K chunks
// [8*hh, 8*hh+8) and [8*hh+16, 8*hh+24) where hh = lane>>4.  rowp points at K=0 of that col.
__device__ __forceinline__ v16bf load_b_frag(const bf16* rowp, int hh) {
  const bf16* q = rowp + 8 * hh;
  v16bf f;
  ((u32x4*)&f)[0] = *(const u32x4*)(q);
  ((u32x4*)&f)[1] = *(const u32x4*)(q + 16);
  return f;
}

// ---------------------------------------------------------------------------
// gfx1250 async copy: per-lane 16B global -> LDS, tracked by ASYNCcnt.
// ---------------------------------------------------------------------------
__device__ __forceinline__ unsigned lds_off_of(const bf16* p) {
  return (unsigned)(unsigned long long)(lds_bf16_t*)p;  // AS3 byte offset
}
__device__ __forceinline__ void async_copy_b128(unsigned lds_off, const void* g) {
  unsigned long long ga = (unsigned long long)(uintptr_t)g;
  asm volatile("global_load_async_to_lds_b128 %0, %1, off"
               :: "v"(lds_off), "v"(ga) : "memory");
}
__device__ __forceinline__ void wait_async0() {
  asm volatile("s_wait_asynccnt 0" ::: "memory");
}
__device__ __forceinline__ void wait_ds0() {
  asm volatile("s_wait_dscnt 0" ::: "memory");
}

// ---------------------------------------------------------------------------
// 1) LayerNorm (fp32 in) -> bf16 out.  One 256-thread block per row of 1024.
// ---------------------------------------------------------------------------
__global__ void layernorm_to_bf16(const float* __restrict__ x,
                                  const float* __restrict__ scale,
                                  const float* __restrict__ bias,
                                  bf16* __restrict__ xb) {
  __shared__ float red[2][8];
  const int row = blockIdx.x;
  const float* xr = x + (size_t)row * CDIM;
  float v[4], s = 0.f, s2 = 0.f;
#pragma unroll
  for (int i = 0; i < 4; ++i) {
    v[i] = xr[threadIdx.x + i * 256];
    s += v[i]; s2 += v[i] * v[i];
  }
#pragma unroll
  for (int m = 16; m >= 1; m >>= 1) {
    s  += __shfl_xor(s,  m, 32);
    s2 += __shfl_xor(s2, m, 32);
  }
  const int wid = threadIdx.x >> 5;
  if ((threadIdx.x & 31) == 0) { red[0][wid] = s; red[1][wid] = s2; }
  __syncthreads();
  s = 0.f; s2 = 0.f;
#pragma unroll
  for (int w = 0; w < 8; ++w) { s += red[0][w]; s2 += red[1][w]; }
  const float mu   = s * (1.0f / CDIM);
  const float var  = s2 * (1.0f / CDIM) - mu * mu;
  const float rstd = rsqrtf(var + LN_EPS);
#pragma unroll
  for (int i = 0; i < 4; ++i) {
    const int c = threadIdx.x + i * 256;
    const float y = (v[i] - mu) * rstd * scale[c] + bias[c];
    xb[(size_t)row * CDIM + c] = (bf16)y;
  }
}

// ---------------------------------------------------------------------------
// 2) Transpose + fp32->bf16 weight conversion: W[rows,cols] -> WT[cols,rows].
// ---------------------------------------------------------------------------
__global__ void transpose_to_bf16(const float* __restrict__ W,
                                  bf16* __restrict__ WT,
                                  int rows, int cols) {
  __shared__ float tile[32][33];
  const int c = blockIdx.x * 32 + threadIdx.x;
  const int r = blockIdx.y * 32 + threadIdx.y;
  tile[threadIdx.y][threadIdx.x] = W[(size_t)r * cols + c];
  __syncthreads();
  const int oc   = blockIdx.y * 32 + threadIdx.x;   // output inner = source row
  const int orow = blockIdx.x * 32 + threadIdx.y;   // output row   = source col
  WT[(size_t)orow * rows + oc] = (bf16)tile[threadIdx.x][threadIdx.y];
}

// ---------------------------------------------------------------------------
// 3) bf16 WMMA GEMM:  C[M,N] = A[M,Kd] @ B[Kd,N]  with B given as BT[N,Kd].
//    Block = 4 waves; each wave owns a 32(M) x 64(N) tile (2x4 accumulators,
//    8 WMMA per K-step).  Templated epilogues.
// ---------------------------------------------------------------------------
enum { MODE_Q = 0, MODE_K = 1, MODE_V = 2, MODE_G = 3, MODE_OUT = 4 };

template <int MODE>
__global__ __launch_bounds__(128)
void gemm_bf16(const bf16* __restrict__ A,    // [M, Kd] row-major
               const bf16* __restrict__ BT,   // [N, Kd] row-major (B transposed)
               const float* __restrict__ bias,
               bf16* __restrict__ outB,
               float* __restrict__ outF,
               int M, int N, int Kd) {
  const int lane = threadIdx.x & 31;
  const int wave = threadIdx.x >> 5;
  const int m0 = blockIdx.y * 128 + wave * 32;
  const int n0 = blockIdx.x * 64;
  const int hh = lane >> 4;
  const int ln = lane & 15;
  const int kb = hh * 8;

  const bf16* ar0 = A + (size_t)(m0 +  0 + ln) * Kd;
  const bf16* ar1 = A + (size_t)(m0 + 16 + ln) * Kd;
  const bf16* bp[4];
#pragma unroll
  for (int c = 0; c < 4; ++c) bp[c] = BT + (size_t)(n0 + c * 16 + ln) * Kd;

  v8f acc[2][4] = {};
  for (int k0 = 0; k0 < Kd; k0 += 32) {
    __builtin_prefetch(ar0 + k0 + 256, 0, 0);   // -> global_prefetch_b8
    v16bf a0 = load_a_frag(ar0 + k0, kb);
    v16bf a1 = load_a_frag(ar1 + k0, kb);
#pragma unroll
    for (int c = 0; c < 4; ++c) {
      v16bf b = load_b_frag(bp[c] + k0, hh);
      acc[0][c] = wmma_bf16(a0, b, acc[0][c]);
      acc[1][c] = wmma_bf16(a1, b, acc[1][c]);
    }
  }

#pragma unroll
  for (int i = 0; i < 2; ++i) {
#pragma unroll
    for (int c = 0; c < 4; ++c) {
      const int n = n0 + c * 16 + ln;
#pragma unroll
      for (int r = 0; r < 8; ++r) {
        const int m = m0 + i * 16 + r + 8 * hh;  // C/D row = r + 8*(lane>>4)
        const float val = acc[i][c][r];
        if (MODE == MODE_Q) {
          outB[(size_t)m * N + n] = (bf16)(val * 0.125f);   // 1/sqrt(64)
        } else if (MODE == MODE_K) {
          outB[(size_t)m * N + n] = (bf16)val;
        } else if (MODE == MODE_V) {
          outB[(size_t)n * M + m] = (bf16)val;              // store V^T [HD, L]
        } else if (MODE == MODE_G) {
          const float z = val + bias[n];
          outF[(size_t)m * N + n] = 1.0f / (1.0f + __expf(-z));
        } else {
          outF[(size_t)m * N + n] = val + bias[n];
        }
      }
    }
  }
}

// ---------------------------------------------------------------------------
// 4) Flash attention: one wave per (16-query tile, head), 64 keys/iteration.
//    K/V tiles staged into LDS with global_load_async_to_lds_b128 (ASYNCcnt),
//    S = Q K^T via WMMA, online softmax (fp32, shfl reductions), P re-laid
//    out through LDS into A-fragments, O += P @ V via WMMA, then /l * gate.
// ---------------------------------------------------------------------------
__global__ __launch_bounds__(32)
void attention_kernel(const bf16* __restrict__ Qb,   // [L, HD] (already *0.125)
                      const bf16* __restrict__ Kb,   // [L, HD]
                      const bf16* __restrict__ Vt,   // [HD, L] (V transposed)
                      const float* __restrict__ Gf,  // [L, HD] sigmoid gate
                      bf16* __restrict__ Ob) {       // [L, HD]
  __shared__ __align__(16) bf16 Klds[64][64];   // [key j][d]     8 KB
  __shared__ __align__(16) bf16 Vlds[64][64];   // [d][key j]     8 KB
  __shared__ __align__(16) bf16 Plds[64][16];   // [key j][q row] 2 KB

  const int lane = threadIdx.x;
  const int h  = blockIdx.y;
  const int q0 = blockIdx.x * 16;
  const int hh = lane >> 4, ln = lane & 15, kb = hh * 8;

  const bf16* qrow = Qb + (size_t)(q0 + ln) * HD + h * DHEAD;
  const v16bf qf0 = load_a_frag(qrow, kb);        // d = 0..31
  const v16bf qf1 = load_a_frag(qrow + 32, kb);   // d = 32..63

  v8f o[4] = {};
  float mrow[8], lrow[8];
#pragma unroll
  for (int r = 0; r < 8; ++r) { mrow[r] = -1e30f; lrow[r] = 0.0f; }

  // Per-lane async-copy assignment: each of 16 steps copies 4 rows x 8 chunks.
  const int lr = lane >> 3;                 // row-within-group 0..3
  const int ch = lane & 7;                  // 16B chunk within a 128B row
  const unsigned kba = lds_off_of(&Klds[0][0]) + (unsigned)(lr * 128 + ch * 16);
  const unsigned vba = lds_off_of(&Vlds[0][0]) + (unsigned)(lr * 128 + ch * 16);
  const bf16* kg = Kb + (size_t)lr * HD + h * DHEAD + ch * 8;          // + (j0+i4)*HD
  const bf16* vg = Vt + (size_t)(h * DHEAD + lr) * L_SEQ + ch * 8;     // + i4*L + j0

  for (int j0 = 0; j0 < L_SEQ; j0 += 64) {
    // ---- async-stage K[j0..j0+63][0..63] and V^T[d][j0..j0+63] into LDS ----
    wait_ds0();                       // prior iteration's LDS reads drained
#pragma unroll
    for (int i = 0; i < 16; ++i) {
      async_copy_b128(kba + (unsigned)(i * 4 * 128),
                      kg + (size_t)(j0 + i * 4) * HD);
      async_copy_b128(vba + (unsigned)(i * 4 * 128),
                      vg + (size_t)(i * 4) * L_SEQ + j0);
    }
    wait_async0();

    // ---- S[16 x 64] = Q_tile @ K_tile^T : 4 col-groups x 2 K-steps --------
    v8f s[4] = {};
#pragma unroll
    for (int g = 0; g < 4; ++g) {
      const bf16* kcol = &Klds[g * 16 + ln][0];
      s[g] = wmma_bf16(qf0, load_b_frag(kcol, hh), s[g]);
      s[g] = wmma_bf16(qf1, load_b_frag(kcol + 32, hh), s[g]);
    }

    // ---- online softmax per row (row = r + 8*hh) ---------------------------
#pragma unroll
    for (int r = 0; r < 8; ++r) {
      float mx = fmaxf(fmaxf(s[0][r], s[1][r]), fmaxf(s[2][r], s[3][r]));
      mx = fmaxf(mx, __shfl_xor(mx, 1, 32));
      mx = fmaxf(mx, __shfl_xor(mx, 2, 32));
      mx = fmaxf(mx, __shfl_xor(mx, 4, 32));
      mx = fmaxf(mx, __shfl_xor(mx, 8, 32));
      const float mnew  = fmaxf(mrow[r], mx);
      const float alpha = __expf(mrow[r] - mnew);
      const float e0 = __expf(s[0][r] - mnew);
      const float e1 = __expf(s[1][r] - mnew);
      const float e2 = __expf(s[2][r] - mnew);
      const float e3 = __expf(s[3][r] - mnew);
      s[0][r] = e0; s[1][r] = e1; s[2][r] = e2; s[3][r] = e3;
      float ps = (e0 + e1) + (e2 + e3);
      ps += __shfl_xor(ps, 1, 32);
      ps += __shfl_xor(ps, 2, 32);
      ps += __shfl_xor(ps, 4, 32);
      ps += __shfl_xor(ps, 8, 32);
      lrow[r] = lrow[r] * alpha + ps;
      mrow[r] = mnew;
      o[0][r] = o[0][r] * alpha;
      o[1][r] = o[1][r] * alpha;
      o[2][r] = o[2][r] * alpha;
      o[3][r] = o[3][r] * alpha;
    }

    // ---- re-layout P: accumulator layout -> A-fragments, via LDS -----------
#pragma unroll
    for (int g = 0; g < 4; ++g) {
      bf16 t[8];
#pragma unroll
      for (int r = 0; r < 8; ++r) t[r] = (bf16)s[g][r];
      *(u32x4*)&Plds[g * 16 + ln][hh * 8] = *(u32x4*)t;   // ds_store_b128
    }
    __syncthreads();
    v16bf pf0, pf1;
#pragma unroll
    for (int e = 0; e < 8; ++e) {
      pf0[e]     = Plds[kb + e][ln];
      pf0[8 + e] = Plds[kb + 16 + e][ln];
      pf1[e]     = Plds[32 + kb + e][ln];
      pf1[8 + e] = Plds[32 + kb + 16 + e][ln];
    }

    // ---- O[16 x 64] += P[16 x 64] @ V_tile[64 x 64] ------------------------
#pragma unroll
    for (int c = 0; c < 4; ++c) {
      const bf16* vcol = &Vlds[c * 16 + ln][0];
      o[c] = wmma_bf16(pf0, load_b_frag(vcol, hh), o[c]);       // keys  0..31
      o[c] = wmma_bf16(pf1, load_b_frag(vcol + 32, hh), o[c]);  // keys 32..63
    }
  }

  // ---- finalize: /l, * gate, store bf16 ------------------------------------
#pragma unroll
  for (int c = 0; c < 4; ++c) {
    const int d = c * 16 + ln;
#pragma unroll
    for (int r = 0; r < 8; ++r) {
      const int i = q0 + r + 8 * hh;
      const float g = Gf[(size_t)i * HD + h * DHEAD + d];
      const float val = o[c][r] / lrow[r] * g;
      Ob[(size_t)i * HD + h * DHEAD + d] = (bf16)val;
    }
  }
}

// ---------------------------------------------------------------------------
// Host-side orchestration
// ---------------------------------------------------------------------------
extern "C" void kernel_launch(void* const* d_in, const int* in_sizes, int n_in,
                              void* d_out, int out_size, void* d_ws, size_t ws_size,
                              hipStream_t stream) {
  const float* seq  = (const float*)d_in[0];
  const float* ln_s = (const float*)d_in[1];
  const float* ln_b = (const float*)d_in[2];
  const float* Wq   = (const float*)d_in[3];
  const float* Wk   = (const float*)d_in[4];
  const float* Wv   = (const float*)d_in[5];
  const float* Wg   = (const float*)d_in[6];
  const float* bg   = (const float*)d_in[7];
  const float* Wo   = (const float*)d_in[8];
  const float* bo   = (const float*)d_in[9];
  float* out = (float*)d_out;

  // Workspace carve-up (~66 MB total)
  char* p = (char*)d_ws;
  bf16* xb  = (bf16*)p;  p += (size_t)L_SEQ * CDIM * 2;
  bf16* WqT = (bf16*)p;  p += (size_t)CDIM * HD * 2;
  bf16* WkT = (bf16*)p;  p += (size_t)CDIM * HD * 2;
  bf16* WvT = (bf16*)p;  p += (size_t)CDIM * HD * 2;
  bf16* WgT = (bf16*)p;  p += (size_t)CDIM * HD * 2;
  bf16* WoT = (bf16*)p;  p += (size_t)HD * CDIM * 2;
  bf16* Qb  = (bf16*)p;  p += (size_t)L_SEQ * HD * 2;
  bf16* Kb  = (bf16*)p;  p += (size_t)L_SEQ * HD * 2;
  bf16* Vtb = (bf16*)p;  p += (size_t)HD * L_SEQ * 2;
  bf16* Ob  = (bf16*)p;  p += (size_t)L_SEQ * HD * 2;
  float* Gf = (float*)p; p += (size_t)L_SEQ * HD * 4;

  // 1) LayerNorm -> bf16 x
  layernorm_to_bf16<<<L_SEQ, 256, 0, stream>>>(seq, ln_s, ln_b, xb);

  // 2) Weight transposes (fp32 -> bf16, [K,N] -> [N,K])
  dim3 tb(32, 32), tg(CDIM / 32, CDIM / 32);
  transpose_to_bf16<<<tg, tb, 0, stream>>>(Wq, WqT, CDIM, HD);
  transpose_to_bf16<<<tg, tb, 0, stream>>>(Wk, WkT, CDIM, HD);
  transpose_to_bf16<<<tg, tb, 0, stream>>>(Wv, WvT, CDIM, HD);
  transpose_to_bf16<<<tg, tb, 0, stream>>>(Wg, WgT, CDIM, HD);
  transpose_to_bf16<<<tg, tb, 0, stream>>>(Wo, WoT, HD, CDIM);

  // 3) Projections (WMMA bf16), 32x64 tile per wave
  dim3 gg(HD / 64, L_SEQ / 128);
  gemm_bf16<MODE_Q><<<gg, 128, 0, stream>>>(xb, WqT, nullptr, Qb,  nullptr, L_SEQ, HD, CDIM);
  gemm_bf16<MODE_K><<<gg, 128, 0, stream>>>(xb, WkT, nullptr, Kb,  nullptr, L_SEQ, HD, CDIM);
  gemm_bf16<MODE_V><<<gg, 128, 0, stream>>>(xb, WvT, nullptr, Vtb, nullptr, L_SEQ, HD, CDIM);
  gemm_bf16<MODE_G><<<gg, 128, 0, stream>>>(xb, WgT, bg,      nullptr, Gf, L_SEQ, HD, CDIM);

  // 4) Flash attention + gate (async LDS staging, WMMA core)
  attention_kernel<<<dim3(L_SEQ / 16, HEADS), 32, 0, stream>>>(Qb, Kb, Vtb, Gf, Ob);

  // 5) Output projection + bias (fp32 out)
  dim3 go(CDIM / 64, L_SEQ / 128);
  gemm_bf16<MODE_OUT><<<go, 128, 0, stream>>>(Ob, WoT, bo, nullptr, out, L_SEQ, CDIM, HD);
}
